// ScaledDotProductAttention_56272661512732
// MI455X (gfx1250) — compile-verified, matrix operand order checked by reading, not run on hardware
//
#include <hip/hip_runtime.h>

// CDNA5 / gfx1250 fused attention:  out = softmax(QK^T/8) V,  also materializes attn.
// B=4 H=16 S=2048 D=64, fp32 I/O, bf16 WMMA compute with f32 accumulate.
// One workgroup (4 wave32) per 16-query-row stripe; score stripe lives in LDS.

typedef __attribute__((ext_vector_type(16))) __bf16    v16bf;
typedef __attribute__((ext_vector_type(8)))  float     v8f;
typedef __attribute__((ext_vector_type(4)))  float     v4f;
typedef __attribute__((ext_vector_type(8)))  unsigned  v8u;
typedef __attribute__((ext_vector_type(4)))  unsigned  v4u;
typedef __attribute__((ext_vector_type(2)))  unsigned  v2u;

#define SEQ   2048
#define HDIM  64
#define LDSTR 2052   // fp32 score row stride: 2048 + 4 pad (16B-aligned, conflict-free)
#define PSTR  2056   // bf16 prob row stride: 2048 + 8 pad (16B-aligned, conflict-free)

// pack two fp32 -> two bf16 in one dword
__device__ __forceinline__ unsigned pk2bf(float lo, float hi) {
#if __has_builtin(__builtin_amdgcn_cvt_pk_bf16_f32)
    return __builtin_bit_cast(unsigned, __builtin_amdgcn_cvt_pk_bf16_f32(lo, hi));
#else
    unsigned ul = __builtin_bit_cast(unsigned, lo) + 0x8000u;   // round-half-up bias
    unsigned uh = __builtin_bit_cast(unsigned, hi) + 0x8000u;
    return __builtin_amdgcn_perm(uh, ul, 0x07060302u);          // {uh[31:16], ul[31:16]}
#endif
}

// Load one 16x32 bf16 WMMA operand row for this lane from a contiguous fp32 row,
// with a compile-time-constant scale folded into the conversion.
// ISA 16-bit A/B striping: element g*8+j  <->  K = g*16 + halfsel*8 + j.
__device__ __forceinline__ v16bf load_row_bf16(const float* __restrict__ p, int halfsel,
                                               float scale) {
    v8u u;
    #pragma unroll
    for (int g = 0; g < 2; ++g) {
        const v4f* q = (const v4f*)(p + g * 16 + halfsel * 8);
        const v4f a = q[0], b = q[1];
        u[g * 4 + 0] = pk2bf(a[0] * scale, a[1] * scale);
        u[g * 4 + 1] = pk2bf(a[2] * scale, a[3] * scale);
        u[g * 4 + 2] = pk2bf(b[0] * scale, b[1] * scale);
        u[g * 4 + 3] = pk2bf(b[2] * scale, b[3] * scale);
    }
    return __builtin_bit_cast(v16bf, u);
}

__global__ __launch_bounds__(128, 1)
void ScaledDotProductAttention_56272661512732_kernel(
    const float* __restrict__ Q, const float* __restrict__ K,
    const float* __restrict__ V, const int* __restrict__ mask,
    float* __restrict__ out, float* __restrict__ attn)
{
    __shared__ float          sc[16 * LDSTR];   // fp32 scores/probs: 131,328 B
    __shared__ unsigned short pb[16 * PSTR];    // bf16 probs:         65,792 B  (total < 320 KB)

    const int tid  = threadIdx.x;
    const int wave = tid >> 5;          // 0..3
    const int lane = tid & 31;          // wave32
    const int bh   = blockIdx.x >> 7;   // / (SEQ/16) q-tiles
    const int qt   = blockIdx.x & 127;
    const int bidx = bh >> 4;           // batch (H = 16)
    const long base = (long)bh * SEQ * HDIM;
    const int q0   = qt * 16;

    const int halfsel = lane >> 4;      // which K-half this lane holds
    const int mrow    = lane & 15;      // row (A) / column (B) owned by this lane

    // -------- Q stripe as two 16x32 bf16 A-operands (1/SCALE pre-folded) --------
    const float* qrow = Q + base + (long)(q0 + mrow) * HDIM;
    const v16bf qa0 = load_row_bf16(qrow,      halfsel, 0.125f);
    const v16bf qa1 = load_row_bf16(qrow + 32, halfsel, 0.125f);

    // ---------------- phase 1: scores = (Q/8)K^T into LDS ----------------
    for (int nt = wave; nt < SEQ / 16; nt += 4) {
        const int n0 = nt * 16;
        const float* krow = K + base + (long)(n0 + mrow) * HDIM;
        const v16bf kb0 = load_row_bf16(krow,      halfsel, 1.0f);
        const v16bf kb1 = load_row_bf16(krow + 32, halfsel, 1.0f);

        v8f acc = {};
        acc = __builtin_amdgcn_wmma_f32_16x16x32_bf16(false, qa0, false, kb0,
                                                      (short)0, acc, false, false);
        acc = __builtin_amdgcn_wmma_f32_16x16x32_bf16(false, qa1, false, kb1,
                                                      (short)0, acc, false, false);
        // C/D layout: element i -> M = i + 8*halfsel, N = mrow
        #pragma unroll
        for (int i = 0; i < 8; ++i) {
            const int m = i + 8 * halfsel;
            sc[m * LDSTR + n0 + mrow] = acc[i];
        }
    }
    __syncthreads();

    // ---------------- phase 2: softmax over each of the 16 rows ----------------
    {
        const int row = wave * 4 + (lane >> 3);   // 8 lanes per row, within one wave
        const int sub = lane & 7;
        const int* __restrict__ mk = mask + (long)bidx * SEQ;

        float mx = -3.402823466e38f;
        for (int c = sub; c < SEQ; c += 8) {
            float v = sc[row * LDSTR + c];
            if (mk[c] == 0) v = -3.402823466e38f;
            mx = fmaxf(mx, v);
        }
        #pragma unroll
        for (int d = 1; d < 8; d <<= 1) mx = fmaxf(mx, __shfl_xor(mx, d, 32));

        float sum = 0.f;
        for (int c = sub; c < SEQ; c += 8) {
            const float v = sc[row * LDSTR + c];
            const float e = (mk[c] == 0) ? 0.f : __expf(v - mx);
            sc[row * LDSTR + c] = e;
            sum += e;
        }
        #pragma unroll
        for (int d = 1; d < 8; d <<= 1) sum += __shfl_xor(sum, d, 32);

        const float inv = 1.0f / sum;
        for (int c = sub; c < SEQ; c += 8) sc[row * LDSTR + c] *= inv;
    }
    __syncthreads();

    // ---- phase 3: stream attn stripe to HBM (coalesced, NT) + bf16 LDS copy ----
    {
        float* arow = attn + (long)bh * SEQ * SEQ + (long)q0 * SEQ;
        #pragma unroll 4
        for (int j = 0; j < 64; ++j) {
            const int idx = j * 512 + tid * 4;       // covers 16*2048 floats
            const int r = idx >> 11;
            const int c = idx & 2047;
            const v4f v = *(const v4f*)&sc[r * LDSTR + c];
            __builtin_nontemporal_store(v, (v4f*)&arow[idx]);
            v2u p;
            p[0] = pk2bf(v[0], v[1]);
            p[1] = pk2bf(v[2], v[3]);
            *(v2u*)&pb[r * PSTR + c] = p;            // 8B-aligned ds_store_b64
        }
    }
    __syncthreads();

    // ------------ phase 4: O = P * V (each wave owns one 16-col slice of D) ------------
    {
        const int n0 = wave * 16;
        const float* vcol = V + base + n0 + mrow;    // V[k][n0+mrow], stride HDIM
        v8f acc = {};
        for (int kc = 0; kc < SEQ / 32; ++kc) {
            // A operand: two 16B LDS vector loads of packed bf16 probs
            const v4u plo = *(const v4u*)&pb[mrow * PSTR + kc * 32 + halfsel * 8];
            const v4u phi = *(const v4u*)&pb[mrow * PSTR + kc * 32 + 16 + halfsel * 8];
            v8u ua;
            #pragma unroll
            for (int i = 0; i < 4; ++i) { ua[i] = plo[i]; ua[4 + i] = phi[i]; }
            const v16bf pa = __builtin_bit_cast(v16bf, ua);

            // B operand: strided gather from V columns, packed to bf16
            v8u uv;
            #pragma unroll
            for (int g = 0; g < 2; ++g) {
                #pragma unroll
                for (int jj = 0; jj < 4; ++jj) {
                    const int k = kc * 32 + g * 16 + halfsel * 8 + jj * 2;
                    const float f0 = vcol[(long)k * HDIM];
                    const float f1 = vcol[(long)(k + 1) * HDIM];
                    uv[g * 4 + jj] = pk2bf(f0, f1);
                }
            }
            const v16bf vb = __builtin_bit_cast(v16bf, uv);

            acc = __builtin_amdgcn_wmma_f32_16x16x32_bf16(false, pa, false, vb,
                                                          (short)0, acc, false, false);
        }
        float* orow = out + base + (long)q0 * HDIM + n0;
        #pragma unroll
        for (int i = 0; i < 8; ++i) {
            const int m = i + 8 * halfsel;
            orow[(long)m * HDIM + mrow] = acc[i];
        }
    }
}

extern "C" void kernel_launch(void* const* d_in, const int* in_sizes, int n_in,
                              void* d_out, int out_size, void* d_ws, size_t ws_size,
                              hipStream_t stream) {
    (void)in_sizes; (void)n_in; (void)out_size; (void)d_ws; (void)ws_size;
    const float* Q    = (const float*)d_in[0];
    const float* K    = (const float*)d_in[1];
    const float* V    = (const float*)d_in[2];
    const int*   mask = (const int*)d_in[3];
    float* out  = (float*)d_out;
    float* attn = out + (size_t)4 * 16 * SEQ * HDIM;   // tuple: (output, attn)

    const int nblocks = 4 * 16 * (SEQ / 16);           // 8192 workgroups of 128 threads
    ScaledDotProductAttention_56272661512732_kernel<<<nblocks, 128, 0, stream>>>(
        Q, K, V, mask, out, attn);
}
